// BatchChildSumTreeLSTM_38302518345967
// MI455X (gfx1250) — matrix operand order repaired and checked by reading, not compiled, not run on hardware
//
#include <hip/hip_runtime.h>
#include <cstdint>
#include <cstddef>

// Problem constants (from reference)
#define B_SZ    1024
#define NPT     127      // 2^(6+1)-1 nodes
#define H_DIM   300      // DIN == H == 300
#define L_OUT   5
#define DEPTH   6

// GEMM packing constants
#define KPAD    640      // K = 600 (x | ch_sum) padded to 640
#define NPACK   1280     // 19 j-blocks * 64 packed cols, padded to 20*64
#define BLK_M   128      // rows per block
#define BLK_NP  256      // packed cols per block (4 j-blocks = 64 cols * 4 gates)
#define KT      32       // K tile per pipeline stage (1 WMMA k-step)
#define NK      (KPAD / KT)     // 20 pipeline stages
#define LDSTR   40       // LDS row stride in bf16 elems (32 + 8 pad)

typedef __attribute__((ext_vector_type(8)))  float        v8f;
typedef __attribute__((ext_vector_type(16))) __bf16       v16bf;
typedef __attribute__((ext_vector_type(4)))  unsigned int v4u;
typedef __attribute__((ext_vector_type(4)))  int          v4i;
typedef __attribute__((ext_vector_type(8)))  int          v8i;

#if defined(__has_builtin)
#if __has_builtin(__builtin_amdgcn_tensor_load_to_lds) && \
    __has_builtin(__builtin_amdgcn_s_wait_tensorcnt)
#define USE_TDM 1
#endif
#endif

struct F2 { float4 a, b; };   // 32 bytes == v16bf

__device__ __forceinline__ unsigned short f2bf(float f) {
  unsigned int u = __float_as_uint(f);
  u += 0x7FFFu + ((u >> 16) & 1u);          // round-to-nearest-even
  return (unsigned short)(u >> 16);
}

__device__ __forceinline__ float sigm(float x) { return 1.0f / (1.0f + __expf(-x)); }

__device__ __forceinline__ v16bf ld_frag(const unsigned short* p0, const unsigned short* p1) {
  F2 t;
  t.a = *(const float4*)p0;
  t.b = *(const float4*)p1;
  return __builtin_bit_cast(v16bf, t);
}

// ---------------------------------------------------------------------------
// One-time packing of gate weights into bf16, layout Wt[c][k], c = 64*jblk +
// 16*g + (j%16), k = [x-proj 0..299 | h-proj 300..599 | zero pad]. Gate order
// g: 0=i, 1=f, 2=o, 3=u. f's h-block is zero (own_h is always zero).
// ---------------------------------------------------------------------------
__global__ __launch_bounds__(256) void pack_w_kernel(
    unsigned short* __restrict__ Wt,
    const float* __restrict__ Wix, const float* __restrict__ Wfx,
    const float* __restrict__ Wox, const float* __restrict__ Wux,
    const float* __restrict__ Wih, const float* __restrict__ Woh,
    const float* __restrict__ Wuh)
{
  int idx = blockIdx.x * 256 + threadIdx.x;          // < NPACK*KPAD
  int cg = idx / KPAD, k = idx % KPAD;
  int jblk = cg >> 6, g = (cg >> 4) & 3, jl = cg & 15;
  int j = jblk * 16 + jl;
  float v = 0.0f;
  if (j < H_DIM) {
    if (k < H_DIM) {
      const float* W = (g == 0) ? Wix : (g == 1) ? Wfx : (g == 2) ? Wox : Wux;
      v = W[k * H_DIM + j];
    } else if (k < 2 * H_DIM && g != 1) {
      const float* W = (g == 0) ? Wih : (g == 2) ? Woh : Wuh;
      v = W[(k - H_DIM) * H_DIM + j];
    }
  }
  Wt[(size_t)cg * KPAD + k] = f2bf(v);
}

__global__ __launch_bounds__(256) void pack_b_kernel(
    float* __restrict__ bpack,
    const float* __restrict__ bix, const float* __restrict__ bih,
    const float* __restrict__ bfx, const float* __restrict__ bfh,
    const float* __restrict__ box, const float* __restrict__ boh,
    const float* __restrict__ bux, const float* __restrict__ buh)
{
  int cg = blockIdx.x * 256 + threadIdx.x;           // < NPACK
  int jblk = cg >> 6, g = (cg >> 4) & 3, jl = cg & 15;
  int j = jblk * 16 + jl;
  float v = 0.0f;
  if (j < H_DIM) {
    if      (g == 0) v = bix[j] + bih[j];
    else if (g == 1) v = bfx[j] + bfh[j];
    else if (g == 2) v = box[j] + boh[j];
    else             v = bux[j] + buh[j];
  }
  bpack[cg] = v;
}

// ---------------------------------------------------------------------------
// One tree level: fused [embedding-gather + sibling h-sum] -> bf16 GEMM
// (M = B*nl, Kpacked = 640, Npacked = 1216) -> in-register LSTM epilogue.
// 8 waves per block: wave grid 2(M) x 4(N); wave tile 64 x 64 packed
// (= 4 gates x 16 real columns). Double-buffered LDS pipeline; B tile is
// copied by the Tensor Data Mover (pad feature reproduces the LDS stride),
// A tile is gathered/summed/converted by the threads.
// ---------------------------------------------------------------------------
__global__ __launch_bounds__(256) void tree_level_gemm(
    float* __restrict__ h_all, float* __restrict__ fc_all,
    const int* __restrict__ word_idx, const float* __restrict__ emb,
    const unsigned short* __restrict__ Wt, const float* __restrict__ bpack,
    int lvl)
{
  __shared__ __align__(16) unsigned short As[2][BLK_M  * LDSTR];
  __shared__ __align__(16) unsigned short Bs[2][BLK_NP * LDSTR];

  const int tid   = threadIdx.x;
  const int lane  = tid & 31;
  const int wv    = tid >> 5;
  const int waveM = wv & 1;     // 0..1
  const int waveN = wv >> 1;    // 0..3
  const int gM = blockIdx.x;
  const int gN = blockIdx.y;

  const int nl = 1 << lvl;
  const int s  = (1 << lvl) - 1;
  const int cs = (1 << (lvl + 1)) - 1;

  // ---- A tile staging: gather embedding / sum children h, fp32 -> bf16 ----
  auto stageA = [&](int kt, int buf) {
    #pragma unroll
    for (int it = tid; it < BLK_M * (KT / 4); it += 256) {
      int r  = it >> 3;             // KT/4 == 8 packs per row
      int k4 = (it & 7) << 2;
      int R  = gM * BLK_M + r;
      int b  = R >> lvl;
      int p  = R & (nl - 1);
      int kg = kt * KT + k4;
      float f0 = 0.f, f1 = 0.f, f2 = 0.f, f3 = 0.f;
      if (kg < H_DIM) {
        int w = word_idx[b * NPT + s + p];
        const float4 v = *(const float4*)(emb + (size_t)w * H_DIM + kg);
        f0 = v.x; f1 = v.y; f2 = v.z; f3 = v.w;
      } else if (kg < 2 * H_DIM && lvl < DEPTH) {
        size_t c0 = ((size_t)b * NPT + cs + 2 * p) * H_DIM + (kg - H_DIM);
        const float4 va = *(const float4*)(h_all + c0);
        const float4 vb = *(const float4*)(h_all + c0 + H_DIM);
        f0 = va.x + vb.x; f1 = va.y + vb.y; f2 = va.z + vb.z; f3 = va.w + vb.w;
      }
      uint2 pk;
      pk.x = (unsigned)f2bf(f0) | ((unsigned)f2bf(f1) << 16);
      pk.y = (unsigned)f2bf(f2) | ((unsigned)f2bf(f3) << 16);
      *(uint2*)(&As[buf][r * LDSTR + k4]) = pk;
    }
  };

  // ---- B tile staging: packed bf16 weights [n][k], stride LDSTR in LDS ----
  auto stageB = [&](int kt, int buf) {
#ifdef USE_TDM
    if (wv == 0) {
      // TDM descriptor: 1-byte elements, 2D tile 64B x 256 rows, source row
      // stride 1280B, LDS padding 16B after every 64B -> 80B LDS row stride.
      unsigned long long ga =
          (unsigned long long)(size_t)(Wt + (size_t)gN * BLK_NP * KPAD + kt * KT);
      unsigned lds = (unsigned)(size_t)(&Bs[buf][0]);
      v4u g0;
      g0[0] = 1u;                                        // count=1 (valid D#)
      g0[1] = lds;                                       // lds_addr
      g0[2] = (unsigned)ga;                              // global_addr[31:0]
      g0[3] = (unsigned)((ga >> 32) & 0x01FFFFFFu) | (2u << 30);  // [56:32] | type=2
      v8i g1;
      g1[0] = (int)((1u << 20) | (3u << 22) | (3u << 25)); // pad_en, intv=16dw, amt=4dw
      g1[1] = (int)((unsigned)(KPAD * 2) << 16);           // tensor_dim0 = 1280 (lo16)
      g1[2] = (int)((unsigned)NPACK << 16);                // tensor_dim1 = 1280 (lo16)
      g1[3] = (int)((unsigned)(KT * 2) << 16);             // tile_dim0 = 64 bytes
      g1[4] = (int)BLK_NP;                                 // tile_dim1 = 256 rows
      g1[5] = (int)(KPAD * 2);                             // tensor_dim0_stride = 1280
      g1[6] = 0;
      g1[7] = 0;
      v4i gz = {0, 0, 0, 0};
#if __clang_major__ >= 23
      v8i gz8 = {0, 0, 0, 0, 0, 0, 0, 0};
      __builtin_amdgcn_tensor_load_to_lds(g0, g1, gz, gz, gz8, 0);
#else
      __builtin_amdgcn_tensor_load_to_lds(g0, g1, gz, gz, 0);
#endif
    }
#else
    #pragma unroll
    for (int it = tid; it < BLK_NP * (KT / 4); it += 256) {
      int n  = it >> 3;
      int k4 = (it & 7) << 2;
      uint2 w4 = *(const uint2*)(Wt + (size_t)(gN * BLK_NP + n) * KPAD + kt * KT + k4);
      *(uint2*)(&Bs[buf][n * LDSTR + k4]) = w4;
    }
#endif
  };

  v8f acc[4][4];
  {
    v8f z = {};
    #pragma unroll
    for (int a = 0; a < 4; ++a)
      #pragma unroll
      for (int g = 0; g < 4; ++g) acc[a][g] = z;
  }

  // ---- software-pipelined main loop (double-buffered LDS) ----
  stageB(0, 0);
  stageA(0, 0);
  for (int kt = 0; kt < NK; ++kt) {
    const int buf = kt & 1;
#ifdef USE_TDM
    if (wv == 0) __builtin_amdgcn_s_wait_tensorcnt(0);  // B tile for kt landed
#endif
    __syncthreads();                                     // tiles[buf] visible to all
    if (kt + 1 < NK) {
      stageB(kt + 1, buf ^ 1);                           // overlap with WMMA below
      stageA(kt + 1, buf ^ 1);
    }
    const unsigned short* Ab = &As[buf][0];
    const unsigned short* Bb = &Bs[buf][0];
    v16bf af[4], bq[4];
    #pragma unroll
    for (int ms = 0; ms < 4; ++ms) {
      int m   = waveM * 64 + ms * 16 + (lane & 15);
      int off = m * LDSTR + ((lane >= 16) ? 8 : 0);
      af[ms] = ld_frag(Ab + off, Ab + off + 16);         // K {b..b+7},{b+16..b+23}
    }
    #pragma unroll
    for (int g = 0; g < 4; ++g) {
      int n   = waveN * 64 + g * 16 + (lane & 15);
      int off = n * LDSTR + ((lane >= 16) ? 8 : 0);
      bq[g] = ld_frag(Bb + off, Bb + off + 16);          // K {b..b+7},{b+16..b+23}
    }
    #pragma unroll
    for (int ms = 0; ms < 4; ++ms)
      #pragma unroll
      for (int g = 0; g < 4; ++g)
        acc[ms][g] = __builtin_amdgcn_wmma_f32_16x16x32_bf16(
            false, af[ms], false, bq[g], (short)0, acc[ms][g], false, false);
  }

  // ---- fused LSTM epilogue: the wave's 4 N-subtiles are gates i,f,o,u ----
  const int jl   = lane & 15;
  const int hi   = lane >> 4;
  const int jblk = gN * 4 + waveN;
  const int j    = jblk * 16 + jl;
  if (j < H_DIM) {
    const int cb = jblk * 64 + jl;
    const float bI = bpack[cb], bF = bpack[cb + 16];
    const float bO = bpack[cb + 32], bU = bpack[cb + 48];
    #pragma unroll
    for (int ms = 0; ms < 4; ++ms) {
      #pragma unroll
      for (int v = 0; v < 8; ++v) {
        int m = waveM * 64 + ms * 16 + v + hi * 8;   // C/D layout: hi half -> M+8
        int R = gM * BLK_M + m;
        int b = R >> lvl;
        int p = R & (nl - 1);
        float gi = sigm(acc[ms][0][v] + bI);
        float gf = sigm(acc[ms][1][v] + bF);
        float go = sigm(acc[ms][2][v] + bO);
        float gu = tanhf(acc[ms][3][v] + bU);
        float cf = 0.0f;
        if (lvl < DEPTH) {
          size_t c0 = ((size_t)b * NPT + cs + 2 * p) * H_DIM + j;
          cf = fc_all[c0] + fc_all[c0 + H_DIM];      // children f*c, fp32
        }
        float c = gi * gu + cf;                      // own_c == 0
        float h = go * tanhf(c);
        size_t o0 = ((size_t)b * NPT + s + p) * H_DIM + j;
        h_all[o0]  = h;
        fc_all[o0] = gf * c;
      }
    }
  }
}

// ---------------------------------------------------------------------------
// Output head: logits = h_root @ W_out + b_out (1024x300 @ 300x5), log_softmax
// ---------------------------------------------------------------------------
__global__ __launch_bounds__(256) void out_logits_kernel(
    const float* __restrict__ h_all, const float* __restrict__ W_out,
    const float* __restrict__ b_out, float* __restrict__ logp)
{
  int b = blockIdx.x * 256 + threadIdx.x;
  if (b >= B_SZ) return;
  const float* h = h_all + (size_t)b * NPT * H_DIM;   // node 0 == root
  float lg[L_OUT];
  #pragma unroll
  for (int c = 0; c < L_OUT; ++c) lg[c] = b_out[c];
  for (int k = 0; k < H_DIM; ++k) {
    float x = h[k];
    const float* wr = W_out + k * L_OUT;
    #pragma unroll
    for (int c = 0; c < L_OUT; ++c) lg[c] = fmaf(x, wr[c], lg[c]);
  }
  float mx = lg[0];
  #pragma unroll
  for (int c = 1; c < L_OUT; ++c) mx = fmaxf(mx, lg[c]);
  float se = 0.0f;
  #pragma unroll
  for (int c = 0; c < L_OUT; ++c) se += __expf(lg[c] - mx);
  float lse = mx + __logf(se);
  #pragma unroll
  for (int c = 0; c < L_OUT; ++c) logp[b * L_OUT + c] = lg[c] - lse;
}

__global__ __launch_bounds__(256) void loss_kernel(
    const float* __restrict__ logp, const int* __restrict__ gold,
    float* __restrict__ out)
{
  __shared__ float red[256];
  int tid = threadIdx.x;
  float sacc = 0.0f;
  for (int b = tid; b < B_SZ; b += 256) sacc -= logp[b * L_OUT + gold[b]];
  red[tid] = sacc;
  __syncthreads();
  for (int off = 128; off > 0; off >>= 1) {
    if (tid < off) red[tid] += red[tid + off];
    __syncthreads();
  }
  if (tid == 0) out[B_SZ * L_OUT] = red[0] / (float)B_SZ;
}

// ---------------------------------------------------------------------------
extern "C" void kernel_launch(void* const* d_in, const int* in_sizes, int n_in,
                              void* d_out, int out_size, void* d_ws, size_t ws_size,
                              hipStream_t stream) {
  (void)in_sizes; (void)n_in; (void)out_size; (void)ws_size;
  const int*   word_idx = (const int*)  d_in[0];
  const int*   gold     = (const int*)  d_in[1];
  const float* emb      = (const float*)d_in[2];
  const float* W_ix = (const float*)d_in[3];  const float* b_ix = (const float*)d_in[4];
  const float* W_ih = (const float*)d_in[5];  const float* b_ih = (const float*)d_in[6];
  const float* W_fx = (const float*)d_in[7];  const float* b_fx = (const float*)d_in[8];
  /* W_fh (d_in[9]) multiplies own_h == 0; only its bias matters */
  const float* b_fh = (const float*)d_in[10];
  const float* W_ox = (const float*)d_in[11]; const float* b_ox = (const float*)d_in[12];
  const float* W_oh = (const float*)d_in[13]; const float* b_oh = (const float*)d_in[14];
  const float* W_ux = (const float*)d_in[15]; const float* b_ux = (const float*)d_in[16];
  const float* W_uh = (const float*)d_in[17]; const float* b_uh = (const float*)d_in[18];
  const float* W_out = (const float*)d_in[19]; const float* b_out = (const float*)d_in[20];

  // Workspace layout (~314 MB): h_all | fc_all | Wt(bf16) | bpack
  size_t nHC = (size_t)B_SZ * NPT * H_DIM;
  float* h_all  = (float*)d_ws;
  float* fc_all = h_all + nHC;
  unsigned short* Wt = (unsigned short*)(fc_all + nHC);
  float* bpack = (float*)(Wt + (size_t)NPACK * KPAD);

  pack_w_kernel<<<(NPACK * KPAD) / 256, 256, 0, stream>>>(
      Wt, W_ix, W_fx, W_ox, W_ux, W_ih, W_oh, W_uh);
  pack_b_kernel<<<NPACK / 256, 256, 0, stream>>>(
      bpack, b_ix, b_ih, b_fx, b_fh, b_ox, b_oh, b_ux, b_uh);

  for (int lvl = DEPTH; lvl >= 0; --lvl) {
    int nl = 1 << lvl;
    dim3 grid((B_SZ * nl) / BLK_M, 5);   // 5 N-blocks cover 19 j-blocks (300 cols)
    tree_level_gemm<<<grid, 256, 0, stream>>>(h_all, fc_all, word_idx, emb,
                                              Wt, bpack, lvl);
  }

  out_logits_kernel<<<B_SZ / 256, 256, 0, stream>>>(h_all, W_out, b_out,
                                                    (float*)d_out);
  loss_kernel<<<1, 256, 0, stream>>>((const float*)d_out, gold, (float*)d_out);
}
